// CausalMultiHeadSelfAttention_1795296330326
// MI455X (gfx1250) — matrix operand order repaired and checked by reading, not compiled, not run on hardware
//
#include <hip/hip_runtime.h>

// ---------------------------------------------------------------------------
// Types for CDNA5 WMMA (wave32, v_wmma_f32_16x16x32_bf16)
// ---------------------------------------------------------------------------
typedef __attribute__((ext_vector_type(16))) __bf16 v16bf;
typedef __attribute__((ext_vector_type(8)))  float  v8f;
typedef __attribute__((ext_vector_type(4)))  unsigned int u32x4;

struct Bytes32 { u32x4 lo, hi; };

static __device__ __forceinline__ u32x4 ld16(const unsigned short* p) {
    u32x4 v;
    __builtin_memcpy(&v, __builtin_assume_aligned(p, 16), 16);
    return v;
}

// A-fragment (16x32 bf16): lane holds 16 contiguous K values (lanes<16: K0..15, lanes>=16: K16..31)
static __device__ __forceinline__ v16bf load_frag_a(const unsigned short* p0,
                                                    const unsigned short* p1) {
    (void)p1;
    Bytes32 b; b.lo = ld16(p0); b.hi = ld16(p0 + 8);
    return __builtin_bit_cast(v16bf, b);
}
// B-fragment (32x16 bf16): lane holds 16 contiguous K values (lanes<16: K0..15, lanes>=16: K16..31)
static __device__ __forceinline__ v16bf load_frag_b(const unsigned short* p) {
    Bytes32 b; b.lo = ld16(p); b.hi = ld16(p + 8);
    return __builtin_bit_cast(v16bf, b);
}

static __device__ __forceinline__ v8f wmma_bf16(v16bf a, v16bf b, v8f c) {
    return __builtin_amdgcn_wmma_f32_16x16x32_bf16(
        /*neg_a=*/false, a, /*neg_b=*/false, b,
        /*c_mod=*/(short)0, c, /*reuse_a=*/false, /*reuse_b=*/false);
}

static __device__ __forceinline__ unsigned short f2bf(float f) {
    unsigned int u = __builtin_bit_cast(unsigned int, f);
    u = (u + 0x7FFFu + ((u >> 16) & 1u)) >> 16;   // round-to-nearest-even
    return (unsigned short)u;
}

// ---------------------------------------------------------------------------
// fp32 -> bf16 convert (grid-stride)
// ---------------------------------------------------------------------------
__global__ void cvt_f32_bf16(const float* __restrict__ in,
                             unsigned short* __restrict__ out, int n) {
    for (int i = blockIdx.x * blockDim.x + threadIdx.x; i < n;
         i += gridDim.x * blockDim.x)
        out[i] = f2bf(in[i]);
}

// ---------------------------------------------------------------------------
// GEMM: out[m,n] = sum_k A[m,k] * B[n,k],  M=8192, N=K=1024 (d)
// B tile (64n x 64k = 8KB) staged into LDS via async copy (ASYNCcnt),
// double-buffered, shared by all 8 waves. 8 WMMAs per stage per wave; all
// fragment loads are batched ahead of the WMMA group so DS latency pipelines.
// mode 0: Q proj + RoPE -> bf16 (b,h,s,dk)
// mode 1: K proj + RoPE -> bf16 (b,h,s,dk)
// mode 2: V proj        -> bf16 (b,h,dk,s)  (transposed for PV B-fragments)
// mode 3: out proj      -> f32  (b,s,d)
// ---------------------------------------------------------------------------
__global__ __launch_bounds__(256) void gemm_bf16_wmma(
    const unsigned short* __restrict__ A,   // M x 1024 bf16 row-major
    const unsigned short* __restrict__ Bm,  // 1024 x 1024 bf16, [n][k]
    float* __restrict__ outF,
    unsigned short* __restrict__ outH,
    const int* __restrict__ pos,            // token positions [2048]
    int mode)
{
    __shared__ __align__(128) unsigned short ldsB[2][64 * 64];  // 2 x 8KB

    const int tid  = threadIdx.x;
    const int lane = tid & 31;
    const int wave = tid >> 5;
    const int lrow = lane & 15;
    const int half = lane >> 4;
    const int m0   = blockIdx.x * 128 + wave * 16;
    const int n0   = blockIdx.y * 64;

    // Async-fill geometry: 8KB tile / 256 threads = two b128 per thread.
    const int o  = tid * 16;         // element offset within 64x64 tile
    const int fn = o >> 6;           // tile row (n)
    const int fk = o & 63;           // k within tile row
    const unsigned short* gBrow = Bm + (size_t)(n0 + fn) * 1024 + fk;

#define ASYNC_FILL(buf, kk)                                                   \
    do {                                                                      \
        unsigned _lds = (unsigned)(uintptr_t)(&ldsB[(buf)][o]);               \
        const unsigned short* _gp = gBrow + (kk);                             \
        asm volatile("global_load_async_to_lds_b128 %0, %1, off"              \
                     :: "v"(_lds), "v"(_gp) : "memory");                      \
        asm volatile("global_load_async_to_lds_b128 %0, %1, off offset:16"    \
                     :: "v"(_lds), "v"(_gp) : "memory");                      \
    } while (0)

    ASYNC_FILL(0, 0);

    v8f zero = {};
    v8f acc[4] = {zero, zero, zero, zero};

    const unsigned short* aRow = A + (size_t)(m0 + lrow) * 1024 + half * 16;

#pragma unroll 1
    for (int kb = 0; kb < 16; ++kb) {
        const int k   = kb * 64;
        const int cur = kb & 1;

        if (kb < 15) {
            ASYNC_FILL(cur ^ 1, k + 64);                 // prefetch next stage
            asm volatile("s_wait_asynccnt 0x2" ::: "memory");
        } else {
            asm volatile("s_wait_asynccnt 0x0" ::: "memory");
        }
        __syncthreads();                                  // all waves' fills of cur landed

        __builtin_prefetch(aRow + k + 128, 0, 3);         // A stream prefetch

        // Batch-load both A fragments and all 8 B fragments, then 8 WMMAs.
        v16bf a[2];
        v16bf bfr[2][4];
#pragma unroll
        for (int c = 0; c < 2; ++c) {
            const unsigned short* ap = aRow + k + c * 32;
            a[c] = load_frag_a(ap, ap + 16);
#pragma unroll
            for (int nt = 0; nt < 4; ++nt)
                bfr[c][nt] = load_frag_b(
                    &ldsB[cur][(nt * 16 + lrow) * 64 + c * 32 + half * 16]);
        }
#pragma unroll
        for (int c = 0; c < 2; ++c)
#pragma unroll
            for (int nt = 0; nt < 4; ++nt)
                acc[nt] = wmma_bf16(a[c], bfr[c][nt], acc[nt]);

        __syncthreads();                                  // cur consumed before refill
    }
#undef ASYNC_FILL

    if (mode == 3) {
#pragma unroll
        for (int nt = 0; nt < 4; ++nt)
#pragma unroll
            for (int r = 0; r < 8; ++r) {
                const int m = m0 + r + half * 8;
                const int n = n0 + nt * 16 + lrow;
                outF[(size_t)m * 1024 + n] = acc[nt][r];
            }
        return;
    }

#pragma unroll
    for (int nt = 0; nt < 4; ++nt) {
        const int n    = n0 + nt * 16 + lrow;
        const int head = n >> 6;
        const int t    = n & 63;
        // freq = 10000^(-(t&~1)/64) = exp(-(t&~1) * ln(10000)/64)
        const float freq = __expf(-(float)(t & ~1) * (9.210340371976184f / 64.0f));
#pragma unroll
        for (int r = 0; r < 8; ++r) {
            const int m    = m0 + r + half * 8;
            const int bidx = m >> 11;
            const int sidx = m & 2047;
            const float val = acc[nt][r];
            if (mode == 2) {
                // V transposed: (b,h,dk,s)
                outH[((size_t)(bidx * 16 + head) * 64 + t) * 2048 + sidx] = f2bf(val);
            } else {
                const float p = (float)pos[sidx];
                float sn, cs;
                __sincosf(p * freq, &sn, &cs);
                const float partner = __shfl_xor(val, 1, 32);  // pair lives in adjacent lane
                const float res = (t & 1) ? (sn * partner + cs * val)
                                          : (cs * val - sn * partner);
                outH[((size_t)(bidx * 16 + head) * 2048 + sidx) * 64 + t] = f2bf(res);
            }
        }
    }
}

// ---------------------------------------------------------------------------
// Causal flash attention.
// Q,K: bf16 (64 bh, 2048 s, 64 dk); Vt: bf16 (64 bh, 64 dk, 2048 s)
// AO : bf16 (4 b, 2048 s, 1024 d)
// One wave: 16 q-rows x full dk. 8 waves/block -> 128 q-rows. Causal loop bound
// is wave-uniform, so EXEC stays all-ones around every WMMA. K and V fragments
// are loaded up front so their global latency overlaps the softmax VALU work.
// ---------------------------------------------------------------------------
__global__ __launch_bounds__(256) void attn_fwd(
    const unsigned short* __restrict__ Q,
    const unsigned short* __restrict__ K,
    const unsigned short* __restrict__ Vt,
    unsigned short* __restrict__ AO)
{
    __shared__ __align__(64) unsigned short plds[8][16 * 32];  // 1KB / wave

    const int lane = threadIdx.x & 31;
    const int wave = threadIdx.x >> 5;
    const int lrow = lane & 15;
    const int half = lane >> 4;
    const int bh   = blockIdx.y;
    const int q0   = blockIdx.x * 128 + wave * 16;

    const unsigned short* Qb = Q  + (size_t)bh * 2048 * 64;
    const unsigned short* Kb = K  + (size_t)bh * 2048 * 64;
    const unsigned short* Vb = Vt + (size_t)bh * 64 * 2048;

    // Q A-fragments: 16x64 split into two 16x32
    const unsigned short* qrow = Qb + (size_t)(q0 + lrow) * 64 + half * 16;
    const v16bf aq0 = load_frag_a(qrow,      qrow + 16);
    const v16bf aq1 = load_frag_a(qrow + 32, qrow + 48);

    v8f zero = {};
    v8f acc[4] = {zero, zero, zero, zero};
    float mI[8], lI[8];
#pragma unroll
    for (int r = 0; r < 8; ++r) { mI[r] = -1e30f; lI[r] = 0.0f; }

    unsigned short* pbase = &plds[wave][0];

    const int jblocks = (q0 + 16 + 31) >> 5;
#pragma unroll 1
    for (int jb = 0; jb < jblocks; ++jb) {
        const int j0 = jb * 32;

        if (jb + 1 < jblocks) {      // prefetch next K/V blocks
            __builtin_prefetch(Kb + (size_t)(j0 + 32 + lrow) * 64, 0, 3);
            __builtin_prefetch(Vb + (size_t)lrow * 2048 + j0 + 32, 0, 3);
        }

        // K B-fragments (S = Q K^T) and V B-fragments (O += P V), loaded
        // together so V's global latency hides under the softmax below.
        v16bf bk[2][2], bv[4];
#pragma unroll
        for (int jt = 0; jt < 2; ++jt)
#pragma unroll
            for (int tb = 0; tb < 2; ++tb)
                bk[jt][tb] = load_frag_b(
                    Kb + (size_t)(j0 + jt * 16 + lrow) * 64 + tb * 32 + half * 16);
#pragma unroll
        for (int nt = 0; nt < 4; ++nt)
            bv[nt] = load_frag_b(
                Vb + (size_t)(nt * 16 + lrow) * 2048 + j0 + half * 16);

        v8f s0 = zero, s1 = zero;
        s0 = wmma_bf16(aq0, bk[0][0], s0);
        s0 = wmma_bf16(aq1, bk[0][1], s0);
        s1 = wmma_bf16(aq0, bk[1][0], s1);
        s1 = wmma_bf16(aq1, bk[1][1], s1);

        const int jg0 = j0 + lrow;
        const int jg1 = j0 + 16 + lrow;
#pragma unroll
        for (int r = 0; r < 8; ++r) {
            const int mg = q0 + r + half * 8;
            float v0 = s0[r] * 0.125f;           // 1/sqrt(64)
            float v1 = s1[r] * 0.125f;
            if (jg0 > mg) v0 = -1e30f;           // causal mask
            if (jg1 > mg) v1 = -1e30f;

            float rm = fmaxf(v0, v1);            // row-max across the 16-lane group
            rm = fmaxf(rm, __shfl_xor(rm, 1, 32));
            rm = fmaxf(rm, __shfl_xor(rm, 2, 32));
            rm = fmaxf(rm, __shfl_xor(rm, 4, 32));
            rm = fmaxf(rm, __shfl_xor(rm, 8, 32));

            const float mnew  = fmaxf(mI[r], rm);
            const float alpha = __expf(mI[r] - mnew);
            mI[r] = mnew;

            const float p0 = __expf(v0 - mnew);
            const float p1 = __expf(v1 - mnew);
            float rs = p0 + p1;
            rs += __shfl_xor(rs, 1, 32);
            rs += __shfl_xor(rs, 2, 32);
            rs += __shfl_xor(rs, 4, 32);
            rs += __shfl_xor(rs, 8, 32);
            lI[r] = lI[r] * alpha + rs;

#pragma unroll
            for (int nt = 0; nt < 4; ++nt) acc[nt][r] *= alpha;

            // P (C-layout) -> LDS row-major 16x32 bf16 (wave-private, DS in-order)
            const int mrow = r + half * 8;
            pbase[mrow * 32 + lrow]      = f2bf(p0);
            pbase[mrow * 32 + 16 + lrow] = f2bf(p1);
        }

        // Reload P as A-fragment (transpose via LDS)
        const unsigned short* pf = pbase + lrow * 32 + half * 16;
        const v16bf pa = load_frag_a(pf, pf + 16);

#pragma unroll
        for (int nt = 0; nt < 4; ++nt)
            acc[nt] = wmma_bf16(pa, bv[nt], acc[nt]);
    }

    // Normalize, write (b, s, h*64+t) bf16 for the final projection GEMM
    const int b = bh >> 4, h = bh & 15;
#pragma unroll
    for (int r = 0; r < 8; ++r) {
        const float inv = 1.0f / lI[r];
        const int sidx  = q0 + r + half * 8;
        unsigned short* orow = AO + (((size_t)(b * 2048 + sidx) * 16) + h) * 64;
#pragma unroll
        for (int nt = 0; nt < 4; ++nt)
            orow[nt * 16 + lrow] = f2bf(acc[nt][r] * inv);
    }
}

// ---------------------------------------------------------------------------
extern "C" void kernel_launch(void* const* d_in, const int* in_sizes, int n_in,
                              void* d_out, int out_size, void* d_ws, size_t ws_size,
                              hipStream_t stream) {
    (void)in_sizes; (void)n_in; (void)out_size; (void)ws_size;

    const float* x  = (const float*)d_in[0];
    const float* Wq = (const float*)d_in[1];
    const float* Wk = (const float*)d_in[2];
    const float* Wv = (const float*)d_in[3];
    const float* Wo = (const float*)d_in[4];
    const int*  pos = (const int*)d_in[5];
    float* out = (float*)d_out;

    const size_t MD = (size_t)4 * 2048 * 1024;  // 8,388,608 activation elems
    const size_t WD = (size_t)1024 * 1024;      // 1,048,576 weight elems

    unsigned char* ws = (unsigned char*)d_ws;
    size_t off = 0;
    auto alloc = [&](size_t bytes) -> void* {
        void* p = ws + off;
        off += (bytes + 255) & ~(size_t)255;
        return p;
    };
    unsigned short* xb  = (unsigned short*)alloc(MD * 2);
    unsigned short* wqb = (unsigned short*)alloc(WD * 2);
    unsigned short* wkb = (unsigned short*)alloc(WD * 2);
    unsigned short* wvb = (unsigned short*)alloc(WD * 2);
    unsigned short* wob = (unsigned short*)alloc(WD * 2);
    unsigned short* Qbf = (unsigned short*)alloc(MD * 2);  // (b,h,s,dk)
    unsigned short* Kbf = (unsigned short*)alloc(MD * 2);  // (b,h,s,dk)
    unsigned short* Vt  = (unsigned short*)alloc(MD * 2);  // (b,h,dk,s)
    unsigned short* AO  = (unsigned short*)alloc(MD * 2);  // (b,s,d)

    // fp32 -> bf16
    cvt_f32_bf16<<<2048, 256, 0, stream>>>(x,  xb,  (int)MD);
    cvt_f32_bf16<<<512,  256, 0, stream>>>(Wq, wqb, (int)WD);
    cvt_f32_bf16<<<512,  256, 0, stream>>>(Wk, wkb, (int)WD);
    cvt_f32_bf16<<<512,  256, 0, stream>>>(Wv, wvb, (int)WD);
    cvt_f32_bf16<<<512,  256, 0, stream>>>(Wo, wob, (int)WD);

    // Projections (+RoPE for Q/K, +transpose for V)
    dim3 gg(64, 16);  // M/128, N/64
    gemm_bf16_wmma<<<gg, 256, 0, stream>>>(xb, wqb, nullptr, Qbf, pos, 0);
    gemm_bf16_wmma<<<gg, 256, 0, stream>>>(xb, wkb, nullptr, Kbf, pos, 1);
    gemm_bf16_wmma<<<gg, 256, 0, stream>>>(xb, wvb, nullptr, Vt,  pos, 2);

    // Causal flash attention
    attn_fwd<<<dim3(16, 64), 256, 0, stream>>>(Qbf, Kbf, Vt, AO);

    // Output projection (fp32 out)
    gemm_bf16_wmma<<<gg, 256, 0, stream>>>(AO, wob, out, nullptr, pos, 3);
}
